// LoRAGate_45346264711435
// MI455X (gfx1250) — matrix operand order repaired and checked by reading, not compiled, not run on hardware
//
#include <hip/hip_runtime.h>
#include <math.h>

// Problem constants (from reference)
#define BTOK   65536
#define DIN    4096
#define NEXP   64
#define TOPK   8
#define NCOL   128     // gate(64) || noise(64)
#define BM     128     // rows per workgroup
#define KC     64      // K-chunk (two 16x16x32 wmma k-steps)
#define NCHUNK (DIN / KC)
#define XPAD   68      // LDS row stride for x tile, in floats (padded)
#define WCHUNK 8192    // packed-w ushorts per K-chunk: 2 ksteps * 8 tiles * 32 lanes * 16

typedef __attribute__((ext_vector_type(16))) __bf16 v16bf;
typedef __attribute__((ext_vector_type(8)))  __bf16 v8bf;
typedef __attribute__((ext_vector_type(8)))  float  v8f;

struct U32x8 { uint4 lo, hi; };
union FragCast { U32x8 q; v16bf v; };
union FragHalves { struct { v8bf lo, hi; } h; v16bf v; };

// ---- gfx1250 async global->LDS copy (probe-confirmed present on this toolchain)
#if defined(__has_builtin)
#if __has_builtin(__builtin_amdgcn_global_load_async_to_lds_b128)
#define HAS_ASYNC 1
#endif
#endif
#ifndef HAS_ASYNC
#define HAS_ASYNC 0
#endif

#if HAS_ASYNC
typedef int v4i_t __attribute__((vector_size(16)));
typedef __attribute__((address_space(1))) v4i_t* gptr_v4i;
typedef __attribute__((address_space(3))) v4i_t* lptr_v4i;

static __device__ __forceinline__ void async_cp16(const void* g, void* l) {
    __builtin_amdgcn_global_load_async_to_lds_b128(
        (gptr_v4i)(void*)g, (lptr_v4i)l, 0, 0);
}
static __device__ __forceinline__ void wait_async() {
#if __has_builtin(__builtin_amdgcn_s_wait_asynccnt)
    __builtin_amdgcn_s_wait_asynccnt(0);
#else
    asm volatile("s_wait_asynccnt 0" ::: "memory");
#endif
}
#else
static __device__ __forceinline__ void wait_async() {}
#endif

// ---- f32x8 -> bf16x8 via vector fptrunc (backend picks native cvt if present)
static __device__ __forceinline__ v8bf cvt8(v8f f) {
    return __builtin_convertvector(f, v8bf);
}

// ---------------------------------------------------------------------------
// Kernel 0: pre-pack [w_gate | w_noise] (f32) -> bf16 in B-fragment lane order.
// Layout: [chunk c][kstep ks][tile n][lane 0..31] -> 16 contiguous bf16 (32 B).
// Lane l: col = n*16 + (l&15); K = c*64 + ks*32 + (l>=16 ? 16 : 0) + 0..15.
// ---------------------------------------------------------------------------
__global__ __launch_bounds__(256) void pack_w(const float* __restrict__ wg,
                                              const float* __restrict__ wn,
                                              ushort* __restrict__ wpack) {
    const int gid  = blockIdx.x * 256 + threadIdx.x;   // 32768 fragments
    const int lane = gid & 31;
    const int n    = (gid >> 5) & 7;
    const int ks   = (gid >> 8) & 1;
    const int c    = gid >> 9;
    const int col  = n * 16 + (lane & 15);
    const int kb   = c * 64 + ks * 32 + ((lane >> 4) ? 16 : 0);
    const float* src = (col < NEXP) ? (wg + col) : (wn + (col - NEXP));
    v8f a, b;
#pragma unroll
    for (int j = 0; j < 8; ++j) {
        a[j] = src[(size_t)(kb + j) * NEXP];
        b[j] = src[(size_t)(kb + 8 + j) * NEXP];
    }
    FragHalves fr;
    fr.h.lo = cvt8(a);
    fr.h.hi = cvt8(b);
    FragCast fc;
    fc.v = fr.v;
    uint4* dst = (uint4*)(wpack + (size_t)gid * 16);
    dst[0] = fc.q.lo;
    dst[1] = fc.q.hi;
}

// ---------------------------------------------------------------------------
// Kernel 1: bf16-WMMA GEMM with ping-pong async-staged LDS buffers.
// logits[B,128] = x[B,4096] @ packed-w
// ---------------------------------------------------------------------------
__global__ __launch_bounds__(256) void gate_gemm(const float* __restrict__ x,
                                                 const ushort* __restrict__ wpack,
                                                 float* __restrict__ logits) {
    __shared__ float  xsf[2][BM * XPAD];   // x tiles, f32 (2 x ~34.8 KB)
    __shared__ ushort wlds[2][WCHUNK];     // packed-w chunks (2 x 16 KB)

    const int tid  = threadIdx.x;
    const int wave = tid >> 5;
    const int lane = tid & 31;
    const int lh   = lane & 15;
    const int hi   = lane >> 4;
    const int rowBase = blockIdx.x * BM;

    v8f acc[8];
#pragma unroll
    for (int n = 0; n < 8; ++n)
#pragma unroll
        for (int j = 0; j < 8; ++j) acc[n][j] = 0.0f;

    // ---- stage one K-chunk into LDS buffer `buf`
    auto stage = [&](int t, int buf) {
        const ushort* wchunk = wpack + (size_t)t * WCHUNK;
        const int k0 = t * KC;
#if HAS_ASYNC
#pragma unroll
        for (int i = 0; i < 8; ++i) {
            int e4 = (tid + i * 256) * 4;
            int r  = e4 >> 6;
            int kk = e4 & 63;
            async_cp16(x + (size_t)(rowBase + r) * DIN + k0 + kk, &xsf[buf][r * XPAD + kk]);
        }
#pragma unroll
        for (int i = 0; i < 4; ++i) {
            int vi = tid + i * 256;
            async_cp16(wchunk + vi * 8, &wlds[buf][vi * 8]);
        }
#else
#pragma unroll
        for (int i = 0; i < 8; ++i) {
            int e4 = (tid + i * 256) * 4;
            int r  = e4 >> 6;
            int kk = e4 & 63;
            *(float4*)&xsf[buf][r * XPAD + kk] =
                *(const float4*)(x + (size_t)(rowBase + r) * DIN + k0 + kk);
        }
#pragma unroll
        for (int i = 0; i < 4; ++i) {
            int vi = tid + i * 256;
            *(uint4*)&wlds[buf][vi * 8] = *(const uint4*)(wchunk + vi * 8);
        }
#endif
    };

    stage(0, 0);  // prologue

    for (int t = 0; t < NCHUNK; ++t) {
        const int cur = t & 1;
        wait_async();       // this wave's copies for chunk t are done
        __syncthreads();    // all waves' copies done; all reads of buf cur^1 done
        if (t + 1 < NCHUNK) stage(t + 1, cur ^ 1);  // overlap DMA with compute

#pragma unroll
        for (int ks = 0; ks < 2; ++ks) {
            const int kb = ks * 32;
            // A fragment: row = wave*16+lh; lanes<16 K{0..15}, lanes>=16 K{16..31}
            const float* ap = &xsf[cur][(wave * 16 + lh) * XPAD + kb + (hi ? 16 : 0)];
            const float4 f0 = *(const float4*)ap;
            const float4 f1 = *(const float4*)(ap + 4);
            const float4 f2 = *(const float4*)(ap + 8);
            const float4 f3 = *(const float4*)(ap + 12);
            v8f fa, fb;
            fa[0] = f0.x; fa[1] = f0.y; fa[2] = f0.z; fa[3] = f0.w;
            fa[4] = f1.x; fa[5] = f1.y; fa[6] = f1.z; fa[7] = f1.w;
            fb[0] = f2.x; fb[1] = f2.y; fb[2] = f2.z; fb[3] = f2.w;
            fb[4] = f3.x; fb[5] = f3.y; fb[6] = f3.z; fb[7] = f3.w;
            FragHalves afr;
            afr.h.lo = cvt8(fa);   // vector fptrunc -> native bf16 cvt if available
            afr.h.hi = cvt8(fb);
            const v16bf A = afr.v;

            // preload all 8 B fragments, then back-to-back WMMAs
            FragCast bfc[8];
#pragma unroll
            for (int n = 0; n < 8; ++n) {
                const ushort* bp = &wlds[cur][(size_t)((ks * 8 + n) * 32 + lane) * 16];
                bfc[n].q.lo = *(const uint4*)bp;
                bfc[n].q.hi = *(const uint4*)(bp + 8);
            }
#pragma unroll
            for (int n = 0; n < 8; ++n) {
                acc[n] = __builtin_amdgcn_wmma_f32_16x16x32_bf16(
                    false, A, false, bfc[n].v, (short)0, acc[n], false, false);
            }
        }
        __syncthreads();  // compute on buf cur finished before its next overwrite
    }

    // write logits per ISA C layout: VGPR r -> M = r (lanes 0-15) / r+8 (lanes 16-31)
#pragma unroll
    for (int n = 0; n < 8; ++n) {
        const int col = n * 16 + lh;
#pragma unroll
        for (int r = 0; r < 8; ++r) {
            const int row = rowBase + wave * 16 + (hi ? r + 8 : r);
            logits[(size_t)row * NCOL + col] = acc[n][r];
        }
    }
}

// ---------------------------------------------------------------------------
// Kernel 2: per-row noisy top-8 sparse softmax + per-row full-softmax stats
// ---------------------------------------------------------------------------
static __device__ __forceinline__ float softplusf(float v) {
    return v > 20.f ? v : log1pf(__expf(v));
}

__global__ __launch_bounds__(256) void gate_topk(const float* __restrict__ logits,
                                                 const float* __restrict__ noise,
                                                 float* __restrict__ out,
                                                 float* __restrict__ rowsp,
                                                 float* __restrict__ rowsq) {
    const int warp = threadIdx.x >> 5;
    const int lane = threadIdx.x & 31;
    const int row  = blockIdx.x * 8 + warp;

    const float2* lrow = (const float2*)(logits + (size_t)row * NCOL);
    const float2 c2 = lrow[lane];
    const float2 n2 = lrow[32 + lane];
    const float2 z2 = ((const float2*)(noise + (size_t)row * NEXP))[lane];

    float v0 = c2.x + z2.x * softplusf(n2.x);
    float v1 = c2.y + z2.y * softplusf(n2.y);

    const float NEG = -__builtin_inff();
    bool s0 = false, s1 = false;
    float m0 = 0.f, sum_sel = 0.f;

#pragma unroll
    for (int it = 0; it < TOPK; ++it) {
        float a0 = s0 ? NEG : v0;
        float a1 = s1 ? NEG : v1;
        float c;  int ci;
        if (a0 >= a1) { c = a0; ci = 2 * lane; } else { c = a1; ci = 2 * lane + 1; }
#pragma unroll
        for (int off = 16; off >= 1; off >>= 1) {
            float oc = __shfl_xor(c, off, 32);
            int   oi = __shfl_xor(ci, off, 32);
            if (oc > c || (oc == c && oi < ci)) { c = oc; ci = oi; }
        }
        if (it == 0) m0 = c;
        sum_sel += __expf(c - m0);
        if (ci == 2 * lane) s0 = true;
        else if (ci == 2 * lane + 1) s1 = true;
    }

    float e0 = __expf(v0 - m0), e1 = __expf(v1 - m0);
    float S = e0 + e1;
#pragma unroll
    for (int off = 16; off >= 1; off >>= 1) S += __shfl_xor(S, off, 32);
    float p0 = e0 / S, p1 = e1 / S;
    float sp = p0 + p1, sq = p0 * p0 + p1 * p1;
#pragma unroll
    for (int off = 16; off >= 1; off >>= 1) {
        sp += __shfl_xor(sp, off, 32);
        sq += __shfl_xor(sq, off, 32);
    }

    float2 o;
    o.x = s0 ? e0 / sum_sel : 0.f;
    o.y = s1 ? e1 / sum_sel : 0.f;
    ((float2*)(out + (size_t)row * NEXP))[lane] = o;

    if (lane == 0) { rowsp[row] = sp; rowsq[row] = sq; }
}

// ---------------------------------------------------------------------------
// Kernel 3: deterministic reduction of row stats -> cv^2 load loss
// ---------------------------------------------------------------------------
__global__ __launch_bounds__(256) void finalize_loss(const float* __restrict__ rowsp,
                                                     const float* __restrict__ rowsq,
                                                     float* __restrict__ out) {
    __shared__ double ssp[256];
    __shared__ double ssq[256];
    const int tid = threadIdx.x;
    double sp = 0.0, sq = 0.0;
    for (int i = tid; i < BTOK; i += 256) { sp += (double)rowsp[i]; sq += (double)rowsq[i]; }
    ssp[tid] = sp; ssq[tid] = sq;
    __syncthreads();
    for (int s = 128; s > 0; s >>= 1) {
        if (tid < s) { ssp[tid] += ssp[tid + s]; ssq[tid] += ssq[tid + s]; }
        __syncthreads();
    }
    if (tid == 0) {
        const double N = (double)BTOK * (double)NEXP;
        double tsp = ssp[0], tsq = ssq[0];
        double mean = tsp / N;
        double var  = (tsq - tsp * tsp / N) / (N - 1.0);
        out[(size_t)BTOK * NEXP] = (float)(var / (mean * mean + 1e-10));
    }
}

// ---------------------------------------------------------------------------
extern "C" void kernel_launch(void* const* d_in, const int* in_sizes, int n_in,
                              void* d_out, int out_size, void* d_ws, size_t ws_size,
                              hipStream_t stream) {
    const float* x  = (const float*)d_in[0];   // [B, D]
    const float* wg = (const float*)d_in[1];   // [D, E]
    const float* wn = (const float*)d_in[2];   // [D, E]
    const float* nz = (const float*)d_in[3];   // [B, E]
    float* out = (float*)d_out;                // [B*E] router_output + [1] load_loss

    float*  wsf    = (float*)d_ws;
    float*  logits = wsf + 64;                         // [B, 128]
    float*  rowsp  = logits + (size_t)BTOK * NCOL;     // [B]
    float*  rowsq  = rowsp + BTOK;                     // [B]
    ushort* wpack  = (ushort*)(rowsq + BTOK);          // [D * 128] bf16, fragment order

    pack_w<<<128, 256, 0, stream>>>(wg, wn, wpack);
    gate_gemm<<<BTOK / BM, 256, 0, stream>>>(x, wpack, logits);
    gate_topk<<<BTOK / 8, 256, 0, stream>>>(logits, nz, out, rowsp, rowsq);
    finalize_loss<<<1, 256, 0, stream>>>(rowsp, rowsq, out);
}
